// MHSA_22600117912207
// MI455X (gfx1250) — matrix operand order repaired and checked by reading, not compiled, not run on hardware
//
#include <hip/hip_runtime.h>

#define IN_DIMC 1024
#define NHC     16
#define DHC     64
#define BATCHC  2
#define SEQC    2048

typedef __attribute__((ext_vector_type(16))) _Float16 v16h;
typedef __attribute__((ext_vector_type(8)))  _Float16 v8h;
typedef __attribute__((ext_vector_type(8)))  float    v8f;

// ---- WMMA fragment loaders (CDNA5 16x16x32 f16 layouts, wave32) ----
// A (16x32, f16): lane holds row (lane&15); halves j=0..7 -> K = k0+8*kg+j,
// halves j=8..15 -> K = k0+16+8*kg+(j-8), with kg = lane>>4.
__device__ __forceinline__ v16h frag_a_f16(const _Float16* __restrict__ base,
                                           int ld, int row0, int k0, int lane) {
  const int r = lane & 15, kg = lane >> 4;
  const _Float16* p = base + (long)(row0 + r) * ld + k0 + 8 * kg;
  v8h lo = *(const v8h*)(p);
  v8h hi = *(const v8h*)(p + 16);
  v16h a;
#pragma unroll
  for (int i = 0; i < 8; ++i) { a[i] = lo[i]; a[8 + i] = hi[i]; }
  return a;
}

// Same A fragment but converting from fp32 source on the fly.
__device__ __forceinline__ v16h frag_a_f32cvt(const float* __restrict__ base,
                                              int ld, int row0, int k0, int lane) {
  const int r = lane & 15, kg = lane >> 4;
  const float* p = base + (long)(row0 + r) * ld + k0 + 8 * kg;
  v16h a;
#pragma unroll
  for (int i = 0; i < 8; ++i) { a[i] = (_Float16)p[i]; a[8 + i] = (_Float16)p[16 + i]; }
  return a;
}

// B (32x16, f16) from an N-major buffer Bt[N][K]: lane holds col n0+(lane&15),
// halves j=0..7 -> K = k0+8*kg+j, j=8..15 -> K = k0+16+8*kg+(j-8), kg = lane>>4.
__device__ __forceinline__ v16h frag_b_nmajor(const _Float16* __restrict__ bt,
                                              long ldk, int n0, long k0, int lane) {
  const _Float16* p = bt + (long)(n0 + (lane & 15)) * ldk + k0 + 8 * (lane >> 4);
  v8h lo = *(const v8h*)(p);
  v8h hi = *(const v8h*)(p + 16);
  v16h b;
#pragma unroll
  for (int i = 0; i < 8; ++i) { b[i] = lo[i]; b[8 + i] = hi[i]; }
  return b;
}

#define WMMA_F32_F16(a, b, c) \
  __builtin_amdgcn_wmma_f32_16x16x32_f16(false, (a), false, (b), (short)0, (c), false, false)

// ---- 1) weight transpose + fp32 -> f16 convert:  out[n*K+k] = in[k*N+n] ----
__global__ void transpose_cvt_kernel(const float* __restrict__ in,
                                     _Float16* __restrict__ out, int K, int N) {
  int idx = blockIdx.x * blockDim.x + threadIdx.x;
  if (idx >= K * N) return;
  int k = idx / N, n = idx - k * N;
  out[(long)n * K + k] = (_Float16)in[idx];
}

// ---- 2) projection GEMM: X[4096x1024]f32 @ Wt(Nmajor)[1024x1024]f16 + bias ----
// one wave = 32 rows x 64 cols (one head); each weight B-fragment feeds 2 WMMAs.
// vtrans=1 stores [b][h][d][s] (for V).
__global__ void proj_kernel(const float* __restrict__ X, const _Float16* __restrict__ Wt,
                            const float* __restrict__ bias, _Float16* __restrict__ dst,
                            int vtrans) {
  const int lane = threadIdx.x & 31;
  const int wv = threadIdx.x >> 5;
  const int m0 = (blockIdx.x * 4 + wv) * 32;
  const int head = blockIdx.y;
  v8f acc[2][4] = {};
  for (int k0 = 0; k0 < IN_DIMC; k0 += 32) {
    v16h a0 = frag_a_f32cvt(X, IN_DIMC, m0, k0, lane);
    v16h a1 = frag_a_f32cvt(X, IN_DIMC, m0 + 16, k0, lane);
#pragma unroll
    for (int t = 0; t < 4; ++t) {
      v16h b = frag_b_nmajor(Wt, IN_DIMC, head * DHC + t * 16, k0, lane);
      acc[0][t] = WMMA_F32_F16(a0, b, acc[0][t]);
      acc[1][t] = WMMA_F32_F16(a1, b, acc[1][t]);
    }
  }
  const int bB = m0 / SEQC;
#pragma unroll
  for (int i = 0; i < 2; ++i) {
#pragma unroll
    for (int t = 0; t < 4; ++t) {
      const int ncol = t * 16 + (lane & 15);
      const float bv = bias[head * DHC + ncol];
#pragma unroll
      for (int vr = 0; vr < 8; ++vr) {
        const int s = (m0 - bB * SEQC) + 16 * i + vr + 8 * (lane >> 4);
        const float val = acc[i][t][vr] + bv;
        long o = vtrans ? ((((long)(bB * NHC + head) * DHC + ncol) * SEQC) + s)
                        : ((((long)(bB * NHC + head) * SEQC + s) * DHC) + ncol);
        dst[o] = (_Float16)val;
      }
    }
  }
}

// ---- 3) attention: one wave = one (b,h) 32-row q tile, loop keys by 32 ----
// sigmoid(x)-0.5 is bounded in (-0.5,0.5) so softmax needs no running max.
__global__ void attn_kernel(const _Float16* __restrict__ qh, const _Float16* __restrict__ kh,
                            const _Float16* __restrict__ vt, _Float16* __restrict__ ctx) {
  __shared__ __align__(64) _Float16 pbuf[4][32][32];
  const int lane = threadIdx.x & 31;
  const int wv = threadIdx.x >> 5;
  const int bh = blockIdx.y;
  const int q0 = (blockIdx.x * 4 + wv) * 32;
  const _Float16* qbh = qh + (long)bh * SEQC * DHC;
  const _Float16* kbh = kh + (long)bh * SEQC * DHC;
  const _Float16* vbh = vt + (long)bh * DHC * SEQC;

  v16h aq[2][2];
#pragma unroll
  for (int i = 0; i < 2; ++i) {
    aq[i][0] = frag_a_f16(qbh, DHC, q0 + 16 * i, 0, lane);
    aq[i][1] = frag_a_f16(qbh, DHC, q0 + 16 * i, 32, lane);
  }

  v8f acc[2][4] = {};
  float rsum[2][8] = {};
  const float scale = 0.125f;  // 1/sqrt(64)

  for (int kk = 0; kk < SEQC; kk += 32) {
    if (kk + 32 < SEQC) {
      __builtin_prefetch(kbh + (long)(kk + 32) * DHC, 0, 1);
      __builtin_prefetch(vbh + (long)(kk + 32), 0, 1);
    }
    // key B-fragments, shared by both q sub-tiles
    v16h bk[2][2];
#pragma unroll
    for (int t = 0; t < 2; ++t) {
      bk[t][0] = frag_b_nmajor(kbh, DHC, kk + t * 16, 0, lane);
      bk[t][1] = frag_b_nmajor(kbh, DHC, kk + t * 16, 32, lane);
    }
#pragma unroll
    for (int i = 0; i < 2; ++i) {
#pragma unroll
      for (int t = 0; t < 2; ++t) {
        v8f sc = {};
        sc = WMMA_F32_F16(aq[i][0], bk[t][0], sc);
        sc = WMMA_F32_F16(aq[i][1], bk[t][1], sc);
#pragma unroll
        for (int vr = 0; vr < 8; ++vr) {
          float x = sc[vr] * scale;
          float sig = 1.0f / (1.0f + __expf(-x));
          float p = __expf(sig - 0.5f);
          rsum[i][vr] += p;
          // C layout: row = vr + 8*(lane>>4), col = lane&15 (within 16-tile t)
          pbuf[wv][16 * i + vr + 8 * (lane >> 4)][t * 16 + (lane & 15)] = (_Float16)p;
        }
      }
    }
    // same-wave LDS RAW: DS pipe is in-order per wave; fence for the scheduler.
    asm volatile("s_wait_dscnt 0" ::: "memory");
    __builtin_amdgcn_wave_barrier();
    v16h pa0 = frag_a_f16(&pbuf[wv][0][0], 32, 0, 0, lane);
    v16h pa1 = frag_a_f16(&pbuf[wv][0][0], 32, 16, 0, lane);
    __builtin_amdgcn_wave_barrier();
#pragma unroll
    for (int dt = 0; dt < 4; ++dt) {
      v16h bv = frag_b_nmajor(vbh, SEQC, dt * 16, kk, lane);  // V^T: contiguous per lane
      acc[0][dt] = WMMA_F32_F16(pa0, bv, acc[0][dt]);
      acc[1][dt] = WMMA_F32_F16(pa1, bv, acc[1][dt]);
    }
  }
  // reduce row sums across the 16 lanes of each half-wave (row group preserved).
#pragma unroll
  for (int i = 0; i < 2; ++i) {
#pragma unroll
    for (int vr = 0; vr < 8; ++vr) {
      float v = rsum[i][vr];
      v += __shfl_xor(v, 1, 32);
      v += __shfl_xor(v, 2, 32);
      v += __shfl_xor(v, 4, 32);
      v += __shfl_xor(v, 8, 32);
      rsum[i][vr] = v;
    }
  }
  const int bB = bh / NHC, h = bh - bB * NHC;
  _Float16* cbase = ctx + (long)bB * SEQC * (NHC * DHC);
#pragma unroll
  for (int i = 0; i < 2; ++i) {
#pragma unroll
    for (int dt = 0; dt < 4; ++dt) {
#pragma unroll
      for (int vr = 0; vr < 8; ++vr) {
        const int row = q0 + 16 * i + vr + 8 * (lane >> 4);
        const int col = h * DHC + dt * 16 + (lane & 15);
        cbase[(long)row * (NHC * DHC) + col] = (_Float16)(acc[i][dt][vr] / rsum[i][vr]);
      }
    }
  }
}

// ---- 4) output projection: ctx[4096x1024]f16 @ WoT[64x1024]f16 + bo -> f32 ----
__global__ void outproj_kernel(const _Float16* __restrict__ ctx, const _Float16* __restrict__ wot,
                               const float* __restrict__ bo, float* __restrict__ out) {
  const int lane = threadIdx.x & 31;
  const int wv = threadIdx.x >> 5;
  const int m0 = (blockIdx.x * 4 + wv) * 32;
  v8f acc[2][4] = {};
  for (int k0 = 0; k0 < NHC * DHC; k0 += 32) {
    v16h a0 = frag_a_f16(ctx, NHC * DHC, m0, k0, lane);
    v16h a1 = frag_a_f16(ctx, NHC * DHC, m0 + 16, k0, lane);
#pragma unroll
    for (int t = 0; t < 4; ++t) {
      v16h b = frag_b_nmajor(wot, NHC * DHC, t * 16, k0, lane);
      acc[0][t] = WMMA_F32_F16(a0, b, acc[0][t]);
      acc[1][t] = WMMA_F32_F16(a1, b, acc[1][t]);
    }
  }
#pragma unroll
  for (int i = 0; i < 2; ++i) {
#pragma unroll
    for (int t = 0; t < 4; ++t) {
      const int n = t * 16 + (lane & 15);
      const float bv = bo[n];
#pragma unroll
      for (int vr = 0; vr < 8; ++vr) {
        const int m = m0 + 16 * i + vr + 8 * (lane >> 4);
        out[(long)m * DHC + n] = acc[i][t][vr] + bv;
      }
    }
  }
}

extern "C" void kernel_launch(void* const* d_in, const int* in_sizes, int n_in,
                              void* d_out, int out_size, void* d_ws, size_t ws_size,
                              hipStream_t stream) {
  const float* q  = (const float*)d_in[0];
  const float* k  = (const float*)d_in[1];
  const float* v  = (const float*)d_in[2];
  const float* Wq = (const float*)d_in[3];
  const float* bq = (const float*)d_in[4];
  const float* Wk = (const float*)d_in[5];
  const float* bk = (const float*)d_in[6];
  const float* Wv = (const float*)d_in[7];
  const float* bv = (const float*)d_in[8];
  const float* Wo = (const float*)d_in[9];
  const float* bo = (const float*)d_in[10];
  float* out = (float*)d_out;

  _Float16* ws = (_Float16*)d_ws;
  size_t off = 0;
  _Float16* wq_t = ws + off; off += (size_t)IN_DIMC * (NHC * DHC);
  _Float16* wk_t = ws + off; off += (size_t)IN_DIMC * (NHC * DHC);
  _Float16* wv_t = ws + off; off += (size_t)IN_DIMC * (NHC * DHC);
  _Float16* wo_t = ws + off; off += (size_t)(NHC * DHC) * DHC;
  _Float16* qh   = ws + off; off += (size_t)BATCHC * NHC * SEQC * DHC;
  _Float16* kh   = ws + off; off += (size_t)BATCHC * NHC * SEQC * DHC;
  _Float16* vt   = ws + off; off += (size_t)BATCHC * NHC * DHC * SEQC;
  _Float16* ctx  = ws + off; off += (size_t)BATCHC * SEQC * NHC * DHC;

  const int n1 = IN_DIMC * NHC * DHC;
  transpose_cvt_kernel<<<(n1 + 255) / 256, 256, 0, stream>>>(Wq, wq_t, IN_DIMC, NHC * DHC);
  transpose_cvt_kernel<<<(n1 + 255) / 256, 256, 0, stream>>>(Wk, wk_t, IN_DIMC, NHC * DHC);
  transpose_cvt_kernel<<<(n1 + 255) / 256, 256, 0, stream>>>(Wv, wv_t, IN_DIMC, NHC * DHC);
  const int n2 = (NHC * DHC) * DHC;
  transpose_cvt_kernel<<<(n2 + 255) / 256, 256, 0, stream>>>(Wo, wo_t, NHC * DHC, DHC);

  dim3 pb(128);                                  // 4 waves per block
  dim3 pg((BATCHC * SEQC / 32) / 4, NHC);        // (32, 16)
  proj_kernel<<<pg, pb, 0, stream>>>(q, wq_t, bq, qh, 0);
  proj_kernel<<<pg, pb, 0, stream>>>(k, wk_t, bk, kh, 0);
  proj_kernel<<<pg, pb, 0, stream>>>(v, wv_t, bv, vt, 1);

  dim3 ag((SEQC / 32) / 4, BATCHC * NHC);        // (16, 32)
  attn_kernel<<<ag, pb, 0, stream>>>(qh, kh, vt, ctx);

  outproj_kernel<<<(BATCHC * SEQC / 32) / 4, pb, 0, stream>>>(ctx, wo_t, bo, out);
}
